// MultiHeadAttention_84241488544067
// MI455X (gfx1250) — matrix operand order repaired and checked by reading, not compiled, not run on hardware
//
#include <hip/hip_runtime.h>

typedef __bf16 bf16_t;
typedef __attribute__((ext_vector_type(8)))  __bf16 v8bf;
typedef __attribute__((ext_vector_type(16))) __bf16 v16bf;
typedef __attribute__((ext_vector_type(8)))  float  v8f;

#define B_  4
#define S_  1024
#define D_  512
#define H_  8
#define DH_ 64

// ---------- helpers ----------

__device__ __forceinline__ bf16_t f2bf(float f) {
  // round-to-nearest-even fp32 -> bf16
  unsigned u = __builtin_bit_cast(unsigned, f);
  u += 0x7FFFu + ((u >> 16) & 1u);
  unsigned short h = (unsigned short)(u >> 16);
  return __builtin_bit_cast(bf16_t, h);
}

__device__ __forceinline__ v16bf load_frag(const bf16_t* p0, const bf16_t* p1) {
  // two contiguous 8-element (16B) chunks -> one 16-lane-element fragment
  v8bf a = *reinterpret_cast<const v8bf*>(p0);
  v8bf b = *reinterpret_cast<const v8bf*>(p1);
  v16bf r;
#pragma unroll
  for (int i = 0; i < 8; ++i) { r[i] = a[i]; r[i + 8] = b[i]; }
  return r;
}

__device__ __forceinline__ float wave_sum(float v) {
#pragma unroll
  for (int off = 16; off > 0; off >>= 1) v += __shfl_xor(v, off, 32);
  return v;
}
__device__ __forceinline__ float wave_max(float v) {
#pragma unroll
  for (int off = 16; off > 0; off >>= 1) v = fmaxf(v, __shfl_xor(v, off, 32));
  return v;
}

__device__ __forceinline__ float fpow_pos(float x, float e) {
  // x**e for x>0, else 0   (v_log_f32 / v_exp_f32 are base-2)
  if (x <= 0.f) return 0.f;
  return __builtin_amdgcn_exp2f(e * __builtin_amdgcn_logf(x));
}

// ---------- kernel 1: fp32 -> bf16 ----------

__global__ void f32_to_bf16_kernel(const float* __restrict__ in,
                                   bf16_t* __restrict__ out, int n) {
  int i = blockIdx.x * blockDim.x + threadIdx.x;
  if (i < n) out[i] = f2bf(in[i]);
}

// ---------- kernel 2: projections  y = X @ W^T + b ----------
// grid.x: (B*S/16) row tiles x (D/16) col tiles; grid.y: matrix 0=q,1=k,2=v
// one wave per block; 16x16 out tile; K loop 512/32 = 16 WMMAs.

__global__ __launch_bounds__(32)
void proj_kernel(const bf16_t* __restrict__ Qb, const bf16_t* __restrict__ Kb,
                 const bf16_t* __restrict__ Vb,
                 const bf16_t* __restrict__ Wqb, const bf16_t* __restrict__ Wkb,
                 const bf16_t* __restrict__ Wvb,
                 const float* __restrict__ bq, const float* __restrict__ bk,
                 const float* __restrict__ bv,
                 bf16_t* __restrict__ qp, bf16_t* __restrict__ kp,
                 bf16_t* __restrict__ vT) {
  const int mat  = blockIdx.y;
  const int jt   = blockIdx.x & 31;   // 32 col tiles over 512
  const int rt   = blockIdx.x >> 5;   // 256 row tiles over B*S
  const int ln   = threadIdx.x;
  const int half = (ln >= 16) ? 1 : 0;
  const int lm   = ln & 15;

  const bf16_t* X    = (mat == 0) ? Qb  : (mat == 1) ? Kb  : Vb;
  const bf16_t* W    = (mat == 0) ? Wqb : (mat == 1) ? Wkb : Wvb;
  const float*  bias = (mat == 0) ? bq  : (mat == 1) ? bk  : bv;

  const int j  = jt * 16 + lm;        // output feature (N column for this lane)
  const float bj = bias[j];
  v8f acc = {bj, bj, bj, bj, bj, bj, bj, bj};

  const int row = rt * 16 + lm;       // A row for this lane
  const bf16_t* Xrow = X + (size_t)row * D_;
  const bf16_t* Wrow = W + (size_t)j * D_;

#pragma unroll 4
  for (int k0 = 0; k0 < D_; k0 += 32) {
    // A (16x32 bf16): lane<16 holds K {k0+0..7, k0+16..23}; lane>=16 +8
    const bf16_t* a0 = Xrow + k0 + (half ? 8 : 0);
    v16bf afrag = load_frag(a0, a0 + 16);
    // B (32x16 bf16): B[k][n] = W[j=n][k]; lane<16 K {k0+0..7, k0+16..23}; lane>=16 +8
    const bf16_t* b0 = Wrow + k0 + (half ? 8 : 0);
    v16bf bfrag = load_frag(b0, b0 + 16);
    acc = __builtin_amdgcn_wmma_f32_16x16x32_bf16(false, afrag, false, bfrag,
                                                  (short)0, acc, false, false);
  }

  const int h = j >> 6, dh = j & 63;
#pragma unroll
  for (int r = 0; r < 8; ++r) {
    const int grow = rt * 16 + r + half * 8;   // C: lanes 0-15 M=r, 16-31 M=r+8
    const int b = grow >> 10;                  // / S_
    const int s = grow & (S_ - 1);
    const size_t bh = (size_t)b * H_ + h;
    const bf16_t val = f2bf(acc[r]);
    if (mat == 2) vT[(bh * DH_ + dh) * S_ + s] = val;        // v transposed
    else {
      bf16_t* dst = (mat == 0) ? qp : kp;
      dst[(bh * S_ + s) * DH_ + dh] = val;                   // [B,H,S,Dh]
    }
  }
}

// ---------- kernel 3: attention (scores + entmax/softmax + PV) ----------
// one block (4 waves / 128 thr) per (b,h,16-query tile).
// LDS: 16x1024 fp32 score stripe (64KB) + 16x1024 bf16 prob stripe (32KB).

__global__ __launch_bounds__(128)
void attn_kernel(const bf16_t* __restrict__ qp, const bf16_t* __restrict__ kp,
                 const bf16_t* __restrict__ vT, const int* __restrict__ seq_mask,
                 const float* __restrict__ alpha_p, const int* __restrict__ sparse_p,
                 float* __restrict__ out) {
  __shared__ __align__(16) float  sc[16 * S_];   // 64 KB: masked scaled scores
  __shared__ __align__(16) bf16_t pb[16 * S_];   // 32 KB: normalized probs (bf16)

  const int ln   = threadIdx.x & 31;
  const int w    = threadIdx.x >> 5;      // wave 0..3
  const int half = (ln >= 16) ? 1 : 0;
  const int lm   = ln & 15;
  const int qt   = blockIdx.x & 63;       // query tile within sequence
  const int bh   = blockIdx.x >> 6;       // b*H + h
  const int b    = bh >> 3;
  const int h    = bh & 7;

  // v stripe this wave needs in phase 3: warm it into cache early
  const int dh_pv = w * 16 + lm;
  const bf16_t* vbase = vT + ((size_t)bh * DH_ + dh_pv) * S_;
  __builtin_prefetch(vbase, 0, 1);
  __builtin_prefetch(vbase + S_ / 2, 0, 1);

  // ---- phase 1: scores = (q k^T)/sqrt(D), masked -> LDS ----
  const size_t qbase = ((size_t)bh * S_ + qt * 16 + lm) * DH_;
  v16bf qa[2];
#pragma unroll
  for (int kk = 0; kk < 2; ++kk) {
    const bf16_t* a0 = qp + qbase + kk * 32 + (half ? 8 : 0);
    qa[kk] = load_frag(a0, a0 + 16);
  }
  const float scale = 0.04419417382415922f;   // 1/sqrt(512)
  const float NEG_INF = -__builtin_inff();

  for (int kt = w; kt < S_ / 16; kt += 4) {
    const int key = kt * 16 + lm;             // N column for this lane
    v8f acc = {0.f, 0.f, 0.f, 0.f, 0.f, 0.f, 0.f, 0.f};
    const bf16_t* kbase = kp + ((size_t)bh * S_ + key) * DH_;
#pragma unroll
    for (int kk = 0; kk < 2; ++kk) {
      const bf16_t* b0 = kbase + kk * 32 + (half ? 8 : 0);
      v16bf bfrag = load_frag(b0, b0 + 16);
      acc = __builtin_amdgcn_wmma_f32_16x16x32_bf16(false, qa[kk], false, bfrag,
                                                    (short)0, acc, false, false);
    }
    const bool valid = (seq_mask[b * S_ + key] != 0);
#pragma unroll
    for (int r = 0; r < 8; ++r) {
      const float v = valid ? acc[r] * scale : NEG_INF;
      sc[(r + half * 8) * S_ + key] = v;
    }
  }
  __syncthreads();

  // ---- phase 2: per-row entmax bisection (or softmax); wave w does rows 4w..4w+3.
  //      Normalized probabilities are written to pb as bf16 for the PV GEMM. ----
  const float alpha   = alpha_p[0];
  const int   sparse  = sparse_p[0];
  const float am1     = alpha - 1.0f;
  const float inv_am1 = 1.0f / am1;
  const bool  sq      = (inv_am1 == 2.0f);    // alpha == 1.5 fast path: pow == square

  for (int rr = 0; rr < 4; ++rr) {
    const int row = w * 4 + rr;
    float xa[32];
    if (sparse != 0) {
      float mx = NEG_INF;
#pragma unroll
      for (int i = 0; i < 32; ++i) {
        xa[i] = sc[row * S_ + ln + 32 * i] * am1;     // Xa = X*(alpha-1)
        mx = fmaxf(mx, xa[i]);
      }
      mx = wave_max(mx);
      float tau_lo = mx - 1.0f;
      const float tau_hi = mx - __builtin_amdgcn_exp2f(-10.0f * am1); // (1/1024)^am1

      auto rowsum = [&](float tau) -> float {
        float s = 0.f;
#pragma unroll
        for (int i = 0; i < 32; ++i) {
          float t = fmaxf(xa[i] - tau, 0.f);
          s += sq ? t * t : fpow_pos(t, inv_am1);
        }
        return wave_sum(s);
      };

      const float f_lo = rowsum(tau_lo) - 1.0f;
      float dm = tau_hi - tau_lo;
      float tau_m = tau_lo;
#pragma unroll 1
      for (int it = 0; it < 50; ++it) {
        dm *= 0.5f;
        tau_m = tau_lo + dm;
        const float f_m = rowsum(tau_m) - 1.0f;
        if (f_m * f_lo >= 0.f) tau_lo = tau_m;
      }
      const float psum = rowsum(tau_m);
      const float inv  = 1.0f / psum;
#pragma unroll
      for (int i = 0; i < 32; ++i) {
        float t = fmaxf(xa[i] - tau_m, 0.f);
        float p = sq ? t * t : fpow_pos(t, inv_am1);
        pb[row * S_ + ln + 32 * i] = f2bf(p * inv);
      }
    } else {
      // softmax
      float mx = NEG_INF;
#pragma unroll
      for (int i = 0; i < 32; ++i) {
        xa[i] = sc[row * S_ + ln + 32 * i];
        mx = fmaxf(mx, xa[i]);
      }
      mx = wave_max(mx);
      float s = 0.f;
#pragma unroll
      for (int i = 0; i < 32; ++i)
        s += __builtin_amdgcn_exp2f((xa[i] - mx) * 1.4426950408889634f);
      s = wave_sum(s);
      const float inv = 1.0f / s;
#pragma unroll
      for (int i = 0; i < 32; ++i)
        pb[row * S_ + ln + 32 * i] =
            f2bf(__builtin_amdgcn_exp2f((xa[i] - mx) * 1.4426950408889634f) * inv);
    }
  }
  __syncthreads();

  // ---- phase 3: out = P @ V ; wave w owns dh tile [16w, 16w+16) ----
  {
    const bf16_t* prow = pb + lm * S_;                  // A row (M = lane%16)
    v8f acc = {0.f, 0.f, 0.f, 0.f, 0.f, 0.f, 0.f, 0.f};
#pragma unroll 4
    for (int k0 = 0; k0 < S_; k0 += 32) {
      // A (16x32 bf16) from LDS probs: lane<16 K {k0+0..7, k0+16..23}; lane>=16 +8
      const bf16_t* a0 = prow + k0 + (half ? 8 : 0);
      v16bf af = load_frag(a0, a0 + 16);
      // B (32x16 bf16) from vT: lane<16 keys {k0+0..7, k0+16..23}; lane>=16 +8
      const bf16_t* b0 = vbase + k0 + (half ? 8 : 0);
      v16bf bfrag = load_frag(b0, b0 + 16);
      acc = __builtin_amdgcn_wmma_f32_16x16x32_bf16(false, af, false, bfrag,
                                                    (short)0, acc, false, false);
    }
    const size_t obase = ((size_t)b * S_ + qt * 16) * D_ + h * DH_ + dh_pv;
#pragma unroll
    for (int r = 0; r < 8; ++r)
      out[obase + (size_t)(r + half * 8) * D_] = acc[r];
  }
}

// ---------- host launcher ----------

extern "C" void kernel_launch(void* const* d_in, const int* in_sizes, int n_in,
                              void* d_out, int out_size, void* d_ws, size_t ws_size,
                              hipStream_t stream) {
  (void)in_sizes; (void)n_in; (void)out_size; (void)ws_size;
  const float* Q        = (const float*)d_in[0];
  const float* K        = (const float*)d_in[1];
  const float* V        = (const float*)d_in[2];
  const int*   seq_mask = (const int*)d_in[3];
  const float* alpha    = (const float*)d_in[4];
  const int*   sparse   = (const int*)d_in[5];
  const float* Wq       = (const float*)d_in[6];
  const float* bq       = (const float*)d_in[7];
  const float* Wk       = (const float*)d_in[8];
  const float* bk       = (const float*)d_in[9];
  const float* Wv       = (const float*)d_in[10];
  const float* bv       = (const float*)d_in[11];
  float*       out      = (float*)d_out;

  char* ws = (char*)d_ws;
  const int NBIG = B_ * S_ * D_;        // 2,097,152 elements
  const int NW   = D_ * D_;             // 262,144 elements
  bf16_t* Qb  = (bf16_t*)(ws + 0);
  bf16_t* Kb  = (bf16_t*)(ws + 4194304);
  bf16_t* Vb  = (bf16_t*)(ws + 8388608);
  bf16_t* qp  = (bf16_t*)(ws + 12582912);
  bf16_t* kp  = (bf16_t*)(ws + 16777216);
  bf16_t* vT  = (bf16_t*)(ws + 20971520);
  bf16_t* Wqb = (bf16_t*)(ws + 25165824);
  bf16_t* Wkb = (bf16_t*)(ws + 25690112);
  bf16_t* Wvb = (bf16_t*)(ws + 26214400);

  f32_to_bf16_kernel<<<(NBIG + 255) / 256, 256, 0, stream>>>(Q, Qb, NBIG);
  f32_to_bf16_kernel<<<(NBIG + 255) / 256, 256, 0, stream>>>(K, Kb, NBIG);
  f32_to_bf16_kernel<<<(NBIG + 255) / 256, 256, 0, stream>>>(V, Vb, NBIG);
  f32_to_bf16_kernel<<<(NW + 255) / 256, 256, 0, stream>>>(Wq, Wqb, NW);
  f32_to_bf16_kernel<<<(NW + 255) / 256, 256, 0, stream>>>(Wk, Wkb, NW);
  f32_to_bf16_kernel<<<(NW + 255) / 256, 256, 0, stream>>>(Wv, Wvb, NW);

  // (B*S/16) * (D/16) = 256*32 = 8192 tiles per matrix, 3 matrices
  proj_kernel<<<dim3(8192, 3), 32, 0, stream>>>(Qb, Kb, Vb, Wqb, Wkb, Wvb,
                                                bq, bk, bv, qp, kp, vT);

  // B*H*(S/16) = 2048 blocks
  attn_kernel<<<2048, 128, 0, stream>>>(qp, kp, vT, seq_mask, alpha, sparse, out);
}
